// GATLayer_1374389534963
// MI455X (gfx1250) — matrix operand order repaired and checked by reading, not compiled, not run on hardware
//
#include <hip/hip_runtime.h>
#include <hip/hip_bf16.h>

// GAT layer (PyG GATConv v1 forward) for MI455X / gfx1250, wave32.
//
// Pipeline (all on `stream`):
//   K0 init      : zero d_out / denom / m_ord; convert W -> f16 in WMMA-B-fragment
//                  swizzled layout (each lane's v16h contiguous -> 1 coalesced 32B load)
//   K1 gemm      : h = x @ W via v_wmma_f32_16x16x32_f16 (f32 accumulate)
//   K2 att       : a_src[n,h] = <h[n,h,:], att_src[h]>, same for a_dst (wave/node)
//   K3 edge1     : alpha = leaky_relu(a_src[s]+a_dst[d]); segment-max via ordered-uint atomicMax
//   K4 edge2     : alpha = exp(alpha - m[d]); segment-sum denom via atomicAdd
//   K5 edge3     : out[d] += (alpha/denom) * h[s]   (wave32 per edge, coalesced 1KB row)
//   K6 final     : out = relu(out + bias)
//
// Workspace layout (bytes, all 256-aligned), total ~68.2 MB:
//   Wh    @ 0          : 256*256 f16 (swizzled) = 131072
//   hbuf  @ 131072     : 50000*256 f32          = 51200000
//   a_src @ 51331072   : 50000*4 f32            = 800000
//   a_dst @ 52131072   : 50000*4 f32            = 800000
//   m_ord @ 52931072   : 50000*4 u32            = 800000
//   denom @ 53731072   : 50000*4 f32            = 800000
//   alpha @ 54531072   : 850000*4 f32           = 13600000   (end = 68131072)

#define N_NODES 50000
#define N_EDGES 800000
#define N_WORK  (N_EDGES + N_NODES)   // edges + self loops
#define IN_FEAT 256
#define HEADS   4
#define OUT_C   64                    // per head
#define MTILES  (N_NODES / 16)        // 3125, exact

typedef __attribute__((ext_vector_type(16))) _Float16 v16h;
typedef __attribute__((ext_vector_type(8)))  float    v8f;

__device__ __forceinline__ unsigned f2ord(float f) {
    unsigned b = __float_as_uint(f);
    return (b & 0x80000000u) ? ~b : (b | 0x80000000u);
}
__device__ __forceinline__ float ord2f(unsigned u) {
    unsigned b = (u & 0x80000000u) ? (u & 0x7FFFFFFFu) : ~u;
    return __uint_as_float(b);
}

// ---------------- K0: init + W -> f16 (B-fragment swizzle) ----------------
// Swizzled layout: fragment (ktile, ntile) occupies 32 lanes * 16 halfs, contiguous.
//   lane = (kr/16)*16 + (n%16),  elem j = kr%16   where kr = k%32
//   Wh[ ((ktile*16 + ntile)*32 + lane)*16 + j ] = W[k][n]
// Contiguous k order shared with K1's A fragment (not the ISA table order): lanes 0-15 hold K=0..15 (2/VGPR),
// lanes 16-31 hold K=16..31, column = lane%16.
__global__ __launch_bounds__(256) void k0_init(const float* __restrict__ W,
                                               _Float16* __restrict__ Wh,
                                               float* __restrict__ out,
                                               float* __restrict__ denom,
                                               unsigned* __restrict__ m_ord) {
    int i = blockIdx.x * 256 + threadIdx.x;
    if (i < N_NODES * 256) out[i] = 0.0f;
    if (i < N_NODES * HEADS) { denom[i] = 0.0f; m_ord[i] = 0u; }
    if (i < IN_FEAT * HEADS * OUT_C) {
        int k = i >> 8, n = i & 255;
        int ktile = k >> 5, kr = k & 31;
        int lane  = ((kr >> 4) << 4) | (n & 15);
        int j     = kr & 15;
        int ntile = n >> 4;
        int swz   = (((ktile << 4) + ntile) * 32 + lane) * 16 + j;
        Wh[swz] = (_Float16)W[i];
    }
}

// ---------------- K1: h = x @ W via WMMA ----------------
// One wave -> one 16x64 output strip. grid = (ceil(3125/8), 4), block = 256 (8 waves).
__global__ __launch_bounds__(256) void k1_gemm(const float* __restrict__ x,
                                               const _Float16* __restrict__ Wh,
                                               float* __restrict__ hbuf) {
    const int lane  = threadIdx.x & 31;
    const int wave  = threadIdx.x >> 5;
    const int mtile = blockIdx.x * 8 + wave;
    if (mtile >= MTILES) return;                 // wave-uniform: EXEC stays all-1s for WMMA
    const int ntile0 = blockIdx.y * 4;           // first of 4 consecutive 16-col tiles

    const int row   = mtile * 16 + (lane & 15);
    const int koff  = (lane >> 4) * 16;          // A-frag K sub-offset per half-wave
    const v16h* wsw = (const v16h*)Wh;           // fragment-swizzled B

    v8f acc[4] = {v8f{}, v8f{}, v8f{}, v8f{}};

    for (int ktile = 0; ktile < IN_FEAT / 32; ++ktile) {
        // A fragment: lane holds row `row`;
        // elems 0..15 = K[kt+koff .. +15]
        v16h a;
        const float* xa = x + (size_t)row * IN_FEAT + ktile * 32 + koff;
        #pragma unroll
        for (int j = 0; j < 8; ++j) {
            a[j]     = (_Float16)xa[j];
            a[j + 8] = (_Float16)xa[j + 8];
        }
        #pragma unroll
        for (int t = 0; t < 4; ++t) {
            // One coalesced 32B load per lane: the whole fragment is 1KB contiguous.
            v16h b = wsw[((ktile << 4) + ntile0 + t) * 32 + lane];
            acc[t] = __builtin_amdgcn_wmma_f32_16x16x32_f16(
                false, a, false, b, (short)0, acc[t], false, false);
        }
    }
    // C/D layout: VGPR r -> row (mtile*16 + r + 8*(lane>=16)), col = (ntile0+t)*16 + (lane&15)
    const int mrow = mtile * 16 + ((lane >> 4) << 3);
    #pragma unroll
    for (int t = 0; t < 4; ++t) {
        #pragma unroll
        for (int r = 0; r < 8; ++r) {
            hbuf[(size_t)(mrow + r) * 256 + (ntile0 + t) * 16 + (lane & 15)] = acc[t][r];
        }
    }
}

// ---------------- K2: per-node attention logits (wave per node) ----------------
__global__ __launch_bounds__(256) void k2_att(const float* __restrict__ hbuf,
                                              const float* __restrict__ att_src,
                                              const float* __restrict__ att_dst,
                                              float* __restrict__ a_src,
                                              float* __restrict__ a_dst) {
    const int node = blockIdx.x * 8 + (threadIdx.x >> 5);
    if (node >= N_NODES) return;
    const int lane = threadIdx.x & 31;
    const int head = lane >> 3;                   // 8 lanes per head (8*8=64 cols)
    const int cofs = (lane & 7) * 8;              // column within head

    const float4* hp = (const float4*)(hbuf + (size_t)node * 256 + lane * 8);
    float4 h0 = hp[0], h1 = hp[1];
    const float* as = att_src + head * OUT_C + cofs;
    const float* ad = att_dst + head * OUT_C + cofs;

    float ss = h0.x*as[0] + h0.y*as[1] + h0.z*as[2] + h0.w*as[3]
             + h1.x*as[4] + h1.y*as[5] + h1.z*as[6] + h1.w*as[7];
    float sd = h0.x*ad[0] + h0.y*ad[1] + h0.z*ad[2] + h0.w*ad[3]
             + h1.x*ad[4] + h1.y*ad[5] + h1.z*ad[6] + h1.w*ad[7];
    #pragma unroll
    for (int m = 1; m < 8; m <<= 1) {
        ss += __shfl_xor(ss, m, 32);
        sd += __shfl_xor(sd, m, 32);
    }
    if ((lane & 7) == 0) {
        a_src[node * HEADS + head] = ss;
        a_dst[node * HEADS + head] = sd;
    }
}

// ---------------- K3: leaky-relu logits + segment max ----------------
__global__ __launch_bounds__(256) void k3_edge1(const int* __restrict__ ei,
                                                const float* __restrict__ a_src,
                                                const float* __restrict__ a_dst,
                                                float* __restrict__ alpha,
                                                unsigned* __restrict__ m_ord) {
    int e = blockIdx.x * 256 + threadIdx.x;
    if (e >= N_WORK) return;
    int s, d;
    if (e < N_EDGES) { s = ei[e]; d = ei[N_EDGES + e]; }
    else             { s = d = e - N_EDGES; }
    #pragma unroll
    for (int h = 0; h < HEADS; ++h) {
        float v = a_src[s * HEADS + h] + a_dst[d * HEADS + h];
        v = (v > 0.0f) ? v : 0.2f * v;            // leaky relu, slope 0.2
        alpha[(size_t)e * HEADS + h] = v;
        atomicMax(&m_ord[d * HEADS + h], f2ord(v));
    }
}

// ---------------- K4: exp(alpha - m) + segment sum ----------------
__global__ __launch_bounds__(256) void k4_edge2(const int* __restrict__ ei,
                                                float* __restrict__ alpha,
                                                const unsigned* __restrict__ m_ord,
                                                float* __restrict__ denom) {
    int e = blockIdx.x * 256 + threadIdx.x;
    if (e >= N_WORK) return;
    int d = (e < N_EDGES) ? ei[N_EDGES + e] : (e - N_EDGES);
    #pragma unroll
    for (int h = 0; h < HEADS; ++h) {
        float m = ord2f(m_ord[d * HEADS + h]);
        float a = __expf(alpha[(size_t)e * HEADS + h] - m);
        alpha[(size_t)e * HEADS + h] = a;
        atomicAdd(&denom[d * HEADS + h], a);
    }
}

// ---------------- K5: weighted scatter-add aggregation (wave per edge) ----------------
__global__ __launch_bounds__(256) void k5_edge3(const int* __restrict__ ei,
                                                const float* __restrict__ hbuf,
                                                const float* __restrict__ alpha,
                                                const float* __restrict__ denom,
                                                float* __restrict__ out) {
    const int e = blockIdx.x * 8 + (threadIdx.x >> 5);
    if (e >= N_WORK) return;
    const int lane = threadIdx.x & 31;
    int s, d;
    if (e < N_EDGES) { s = ei[e]; d = ei[N_EDGES + e]; }
    else             { s = d = e - N_EDGES; }
    const int head = lane >> 3;
    const float coef = alpha[(size_t)e * HEADS + head] /
                       (denom[d * HEADS + head] + 1e-16f);

    const float4* hp = (const float4*)(hbuf + (size_t)s * 256 + lane * 8);
    float4 h0 = hp[0], h1 = hp[1];
    float* op = out + (size_t)d * 256 + lane * 8;
    atomicAdd(op + 0, coef * h0.x);
    atomicAdd(op + 1, coef * h0.y);
    atomicAdd(op + 2, coef * h0.z);
    atomicAdd(op + 3, coef * h0.w);
    atomicAdd(op + 4, coef * h1.x);
    atomicAdd(op + 5, coef * h1.y);
    atomicAdd(op + 6, coef * h1.z);
    atomicAdd(op + 7, coef * h1.w);
}

// ---------------- K6: bias + relu ----------------
__global__ __launch_bounds__(256) void k6_final(float* __restrict__ out,
                                                const float* __restrict__ bias) {
    int i = blockIdx.x * 256 + threadIdx.x;
    if (i >= N_NODES * 256) return;
    float v = out[i] + bias[i & 255];
    out[i] = (v > 0.0f) ? v : 0.0f;
}

extern "C" void kernel_launch(void* const* d_in, const int* in_sizes, int n_in,
                              void* d_out, int out_size, void* d_ws, size_t ws_size,
                              hipStream_t stream) {
    const float* x       = (const float*)d_in[0];   // [50000, 256]
    const int*   ei      = (const int*)d_in[1];     // [2, 800000]
    const float* W       = (const float*)d_in[2];   // [256, 256]
    const float* att_src = (const float*)d_in[3];   // [4, 64]
    const float* att_dst = (const float*)d_in[4];   // [4, 64]
    const float* bias    = (const float*)d_in[5];   // [256]
    float* out = (float*)d_out;

    char* ws = (char*)d_ws;
    _Float16* Wh    = (_Float16*)(ws + 0);
    float*    hbuf  = (float*)(ws + 131072);
    float*    a_src = (float*)(ws + 51331072);
    float*    a_dst = (float*)(ws + 52131072);
    unsigned* m_ord = (unsigned*)(ws + 52931072);
    float*    denom = (float*)(ws + 53731072);
    float*    alpha = (float*)(ws + 54531072);

    // K0: init (covers the largest range: N*256)
    k0_init<<<(N_NODES * 256 + 255) / 256, 256, 0, stream>>>(W, Wh, out, denom, m_ord);

    // K1: GEMM, 8 waves/block, each wave = one 16x64 strip
    dim3 ggrid((MTILES + 7) / 8, (HEADS * OUT_C) / 64);
    k1_gemm<<<ggrid, 256, 0, stream>>>(x, Wh, hbuf);

    // K2: attention logits, 8 nodes/block
    k2_att<<<(N_NODES + 7) / 8, 256, 0, stream>>>(hbuf, att_src, att_dst, a_src, a_dst);

    // K3/K4: edge softmax passes
    k3_edge1<<<(N_WORK + 255) / 256, 256, 0, stream>>>(ei, a_src, a_dst, alpha, m_ord);
    k4_edge2<<<(N_WORK + 255) / 256, 256, 0, stream>>>(ei, alpha, m_ord, denom);

    // K5: aggregation, 8 edges/block (wave per edge)
    k5_edge3<<<(N_WORK + 7) / 8, 256, 0, stream>>>(ei, hbuf, alpha, denom, out);

    // K6: bias + relu
    k6_final<<<(N_NODES * 256 + 255) / 256, 256, 0, stream>>>(out, bias);
}
